// MoE_66571993088401
// MI455X (gfx1250) — matrix operand order repaired and checked by reading, not compiled, not run on hardware
//
#include <hip/hip_runtime.h>
#include <math.h>

// ---------------------------------------------------------------------------
// Hard-gumbel top-1 MoE of GLU experts, routed (argmax gate => compute only
// the winning expert per token).  bf16 WMMA (fp32 accum) + CDNA5 async
// global->LDS tile DMA (ASYNCcnt) with double buffering.
//
// Workspace layout (compile-only environment; assumes ws_size ~ 300 MB):
//   [0,256)        counts[3]           (memset 0 per launch)
//   idx            per-expert token index lists      3*N*4 B
//   W1t/W3t/W2t    bf16 weights transposed to [e][n][k]   3*6 MB
//   xb             bf16 copy of x [N][H]             128 MB
//   innerb         bf16 inner activations [N][H]     128 MB
// ---------------------------------------------------------------------------

#define N_TOK 65536
#define H_DIM 1024
#define N_EXP 3

#define BM 128
#define BN 64
#define BK 32
#define KITER (H_DIM / BK)

typedef __attribute__((ext_vector_type(16))) __bf16 v16bf;
typedef __attribute__((ext_vector_type(8)))  __bf16 v8bf;
typedef __attribute__((ext_vector_type(8)))  float  v8f;

// --- CDNA5 async tile DMA helpers (CDNA5 ISA) --------
__device__ __forceinline__ unsigned lds_off(const void* p) {
  // generic pointer to __shared__ = SHARED_BASE aperture | LDS byte offset
  return (unsigned)(unsigned long long)p;
}
__device__ __forceinline__ void async_b128(unsigned lds, const void* g) {
  asm volatile("global_load_async_to_lds_b128 %0, %1, off"
               :: "v"(lds), "v"((unsigned long long)g) : "memory");
}
__device__ __forceinline__ void wait_async0() {
  asm volatile("s_wait_asynccnt 0x0" ::: "memory");
}

// ---------------------------------------------------------------------------
// Kernel 1: fp32 [e][d][h] -> bf16 [e][h][d]  (transpose so each WMMA B
// fragment / async copy is contiguous-in-k).
// ---------------------------------------------------------------------------
__global__ void cvt_transpose_w(const float* __restrict__ src,
                                __bf16* __restrict__ dst) {
  const long long total = (long long)N_EXP * H_DIM * H_DIM;
  for (long long i = (long long)blockIdx.x * blockDim.x + threadIdx.x;
       i < total; i += (long long)gridDim.x * blockDim.x) {
    long long e = i / ((long long)H_DIM * H_DIM);
    int r = (int)(i - e * H_DIM * H_DIM);
    int h = r / H_DIM;
    int d = r % H_DIM;
    dst[i] = (__bf16)src[e * H_DIM * H_DIM + (long long)d * H_DIM + h];
  }
}

// ---------------------------------------------------------------------------
// Kernel 2: gate logits + gumbel + argmax; compacts per-expert token lists
// and (fused, since x is streamed anyway) converts x to bf16 for the GEMMs.
// One wave32 per token.
// ---------------------------------------------------------------------------
__global__ __launch_bounds__(256) void gate_route(
    const float* __restrict__ x, const float* __restrict__ Wg,
    const float* __restrict__ bg, const float* __restrict__ gum,
    int* __restrict__ counts, int* __restrict__ idx,
    __bf16* __restrict__ xb) {
  int tok  = (int)((blockIdx.x * blockDim.x + threadIdx.x) >> 5);
  int lane = threadIdx.x & 31;
  if (tok >= N_TOK) return;
  const float* xr = x + (long long)tok * H_DIM;
  __bf16* xbr = xb + (long long)tok * H_DIM;
  float a0 = 0.f, a1 = 0.f, a2 = 0.f;
  for (int i = lane; i < H_DIM; i += 32) {
    float xv = xr[i];
    xbr[i] = (__bf16)xv;  // fused fp32->bf16 copy of x
    a0 = fmaf(xv, Wg[i * 3 + 0], a0);
    a1 = fmaf(xv, Wg[i * 3 + 1], a1);
    a2 = fmaf(xv, Wg[i * 3 + 2], a2);
  }
  for (int off = 16; off > 0; off >>= 1) {
    a0 += __shfl_xor(a0, off, 32);
    a1 += __shfl_xor(a1, off, 32);
    a2 += __shfl_xor(a2, off, 32);
  }
  if (lane == 0) {
    float l[3] = {a0 + bg[0], a1 + bg[1], a2 + bg[2]};
    int best = 0;
    float bv = -3.4e38f;
    for (int e = 0; e < 3; ++e) {
      float u = gum[tok * 3 + e];
      u = fminf(fmaxf(u, 1e-6f), 1.0f - 1e-6f);
      float g = -__logf(-__logf(u));
      float v = l[e] + g;
      if (v > bv) { bv = v; best = e; }  // first-max-wins == jnp.argmax
    }
    int pos = atomicAdd(&counts[best], 1);
    idx[best * N_TOK + pos] = tok;
  }
}

// ---------------------------------------------------------------------------
// Fragment helpers (wave32; contiguous-k order of this code, not the ISA table order):
//  A (16x32 bf16): lane&15 = M row; 16 contiguous K per lane: K[(lane>>4)*16..+16)
//  B (32x16 bf16): lane&15 = N col; 16 contiguous K per lane from [n][k] tile
// ---------------------------------------------------------------------------
__device__ __forceinline__ v16bf frag_A(const __bf16* base, int row, int hi) {
  const __bf16* p = base + row * BK + hi * 16;
  v8bf lo  = *(const v8bf*)(p);
  v8bf hi8 = *(const v8bf*)(p + 8);
  v16bf r;
#pragma unroll
  for (int q = 0; q < 8; ++q) { r[q] = lo[q]; r[q + 8] = hi8[q]; }
  return r;
}
__device__ __forceinline__ v16bf frag_B(const __bf16* base, int n, int hi) {
  return *(const v16bf*)(base + n * BK + hi * 16);
}

// ---------------------------------------------------------------------------
// Kernel 3: grouped up-projection GLU:
//   inner[tok,n] = act_e(xb@W1e + b1e) * (xb@W3e + b3e)   (bf16 out)
// 256 threads / 8 waves, tile 128x64xBK32, async double-buffered staging.
// ---------------------------------------------------------------------------
__global__ __launch_bounds__(256) void glu_up(
    const __bf16* __restrict__ xb,
    const __bf16* __restrict__ W1t, const __bf16* __restrict__ W3t,
    const float* __restrict__ b1, const float* __restrict__ b3,
    const int* __restrict__ counts, const int* __restrict__ idx,
    __bf16* __restrict__ innerb) {
  __shared__ __attribute__((aligned(32))) __bf16 xt[2][BM * BK];
  __shared__ __attribute__((aligned(32))) __bf16 w1s[2][BN * BK];
  __shared__ __attribute__((aligned(32))) __bf16 w3s[2][BN * BK];
  __shared__ int tokL[BM];

  const int e   = blockIdx.z;
  const int cnt = counts[e];
  const int rb  = blockIdx.y * BM;
  if (rb >= cnt) return;
  const int n0  = blockIdx.x * BN;
  const int tid = threadIdx.x;

  if (tid < BM) {
    int pos = rb + tid;
    tokL[tid] = (pos < cnt) ? idx[e * N_TOK + pos] : -1;
  }
  __syncthreads();

  const int lane = tid & 31, wv = tid >> 5;
  const int wm = wv >> 1, wn = wv & 1;      // 4x2 wave grid of 32x32 subtiles
  const int l16 = lane & 15, hi = lane >> 4;

  const __bf16* W1e = W1t + (long long)e * H_DIM * H_DIM;
  const __bf16* W3e = W3t + (long long)e * H_DIM * H_DIM;

  // per-thread async-staging coordinates (16B granules)
  const int arow = tid >> 2, aseg = tid & 3;        // xt: rows  0..127 (x2)
  const int wrow = tid >> 2, wseg = tid & 3;        // w tiles: n 0..63
  int tokA0 = tokL[arow];        if (tokA0 < 0) tokA0 = tokL[0];
  int tokA1 = tokL[arow + 64];   if (tokA1 < 0) tokA1 = tokL[0];

  auto issue = [&](int k0, int b) {
    // A tile: 128 rows x 32 k  = 512 x 16B  (2 per thread)
    async_b128(lds_off(&xt[b][arow * BK + aseg * 8]),
               xb + (long long)tokA0 * H_DIM + k0 + aseg * 8);
    async_b128(lds_off(&xt[b][(arow + 64) * BK + aseg * 8]),
               xb + (long long)tokA1 * H_DIM + k0 + aseg * 8);
    // W tiles: 64 n x 32 k each = 256 x 16B (1 per thread per matrix)
    async_b128(lds_off(&w1s[b][wrow * BK + wseg * 8]),
               W1e + (long long)(n0 + wrow) * H_DIM + k0 + wseg * 8);
    async_b128(lds_off(&w3s[b][wrow * BK + wseg * 8]),
               W3e + (long long)(n0 + wrow) * H_DIM + k0 + wseg * 8);
  };

  v8f acc1[2][2] = {};
  v8f acc3[2][2] = {};

  issue(0, 0);
  for (int i = 0; i < KITER; ++i) {
    const int b = i & 1;
    wait_async0();        // this wave's copies into buf b have landed
    __syncthreads();      // everyone's copies landed; prev reads of buf b^1 done
    if (i + 1 < KITER) issue((i + 1) * BK, b ^ 1);  // DMA overlaps WMMA below

    v16bf a[2], f1[2], f3[2];
#pragma unroll
    for (int mt = 0; mt < 2; ++mt)
      a[mt] = frag_A(xt[b], wm * 32 + mt * 16 + l16, hi);
#pragma unroll
    for (int nt = 0; nt < 2; ++nt) {
      f1[nt] = frag_B(w1s[b], wn * 32 + nt * 16 + l16, hi);
      f3[nt] = frag_B(w3s[b], wn * 32 + nt * 16 + l16, hi);
    }
#pragma unroll
    for (int mt = 0; mt < 2; ++mt)
#pragma unroll
      for (int nt = 0; nt < 2; ++nt) {
        acc1[mt][nt] = __builtin_amdgcn_wmma_f32_16x16x32_bf16(
            false, a[mt], false, f1[nt], (short)0, acc1[mt][nt], false, false);
        acc3[mt][nt] = __builtin_amdgcn_wmma_f32_16x16x32_bf16(
            false, a[mt], false, f3[nt], (short)0, acc3[mt][nt], false, false);
      }
  }

  // epilogue: bias + activation + elementwise product -> bf16 inner
  // C/D layout: lanes 0-15: (M=r, N=lane); lanes 16-31: (M=r+8, N=lane-16)
#pragma unroll
  for (int mt = 0; mt < 2; ++mt)
#pragma unroll
    for (int nt = 0; nt < 2; ++nt)
#pragma unroll
      for (int r = 0; r < 8; ++r) {
        int ml = wm * 32 + mt * 16 + r + hi * 8;
        int n  = n0 + wn * 32 + nt * 16 + l16;
        int tok = tokL[ml];
        if (tok < 0) continue;
        float h1v = acc1[mt][nt][r] + b1[e * H_DIM + n];
        float h3v = acc3[mt][nt][r] + b3[e * H_DIM + n];
        float act;
        if (e == 0)       act = h1v / (1.f + __expf(-h1v));                           // silu
        else if (e == 1)  act = 0.5f * h1v * (1.f + erff(h1v * 0.7071067811865475f)); // exact gelu
        else              act = fmaxf(h1v, 0.f);                                      // relu
        innerb[(long long)tok * H_DIM + n] = (__bf16)(act * h3v);
      }
}

// ---------------------------------------------------------------------------
// Kernel 4: grouped down-projection:  out[tok,n] = inner@W2e + b2e  (fp32)
// ---------------------------------------------------------------------------
__global__ __launch_bounds__(256) void glu_down(
    const __bf16* __restrict__ innerb,
    const __bf16* __restrict__ W2t, const float* __restrict__ b2,
    const int* __restrict__ counts, const int* __restrict__ idx,
    float* __restrict__ out) {
  __shared__ __attribute__((aligned(32))) __bf16 at[2][BM * BK];
  __shared__ __attribute__((aligned(32))) __bf16 w2s[2][BN * BK];
  __shared__ int tokL[BM];

  const int e   = blockIdx.z;
  const int cnt = counts[e];
  const int rb  = blockIdx.y * BM;
  if (rb >= cnt) return;
  const int n0  = blockIdx.x * BN;
  const int tid = threadIdx.x;

  if (tid < BM) {
    int pos = rb + tid;
    tokL[tid] = (pos < cnt) ? idx[e * N_TOK + pos] : -1;
  }
  __syncthreads();

  const int lane = tid & 31, wv = tid >> 5;
  const int wm = wv >> 1, wn = wv & 1;
  const int l16 = lane & 15, hi = lane >> 4;

  const __bf16* W2e = W2t + (long long)e * H_DIM * H_DIM;

  const int arow = tid >> 2, aseg = tid & 3;
  int tokA0 = tokL[arow];      if (tokA0 < 0) tokA0 = tokL[0];
  int tokA1 = tokL[arow + 64]; if (tokA1 < 0) tokA1 = tokL[0];

  auto issue = [&](int k0, int b) {
    async_b128(lds_off(&at[b][arow * BK + aseg * 8]),
               innerb + (long long)tokA0 * H_DIM + k0 + aseg * 8);
    async_b128(lds_off(&at[b][(arow + 64) * BK + aseg * 8]),
               innerb + (long long)tokA1 * H_DIM + k0 + aseg * 8);
    async_b128(lds_off(&w2s[b][arow * BK + aseg * 8]),
               W2e + (long long)(n0 + arow) * H_DIM + k0 + aseg * 8);
  };

  v8f acc[2][2] = {};

  issue(0, 0);
  for (int i = 0; i < KITER; ++i) {
    const int b = i & 1;
    wait_async0();
    __syncthreads();
    if (i + 1 < KITER) issue((i + 1) * BK, b ^ 1);

    v16bf a[2], f2[2];
#pragma unroll
    for (int mt = 0; mt < 2; ++mt)
      a[mt] = frag_A(at[b], wm * 32 + mt * 16 + l16, hi);
#pragma unroll
    for (int nt = 0; nt < 2; ++nt)
      f2[nt] = frag_B(w2s[b], wn * 32 + nt * 16 + l16, hi);
#pragma unroll
    for (int mt = 0; mt < 2; ++mt)
#pragma unroll
      for (int nt = 0; nt < 2; ++nt)
        acc[mt][nt] = __builtin_amdgcn_wmma_f32_16x16x32_bf16(
            false, a[mt], false, f2[nt], (short)0, acc[mt][nt], false, false);
  }

#pragma unroll
  for (int mt = 0; mt < 2; ++mt)
#pragma unroll
    for (int nt = 0; nt < 2; ++nt)
#pragma unroll
      for (int r = 0; r < 8; ++r) {
        int ml = wm * 32 + mt * 16 + r + hi * 8;
        int n  = n0 + wn * 32 + nt * 16 + l16;
        int tok = tokL[ml];
        if (tok < 0) continue;
        out[(long long)tok * H_DIM + n] = acc[mt][nt][r] + b2[e * H_DIM + n];
      }
}

// ---------------------------------------------------------------------------
extern "C" void kernel_launch(void* const* d_in, const int* in_sizes, int n_in,
                              void* d_out, int out_size, void* d_ws,
                              size_t ws_size, hipStream_t stream) {
  const float* x  = (const float*)d_in[0];
  const float* W1 = (const float*)d_in[1];
  const float* b1 = (const float*)d_in[2];
  const float* W2 = (const float*)d_in[3];
  const float* b2 = (const float*)d_in[4];
  const float* W3 = (const float*)d_in[5];
  const float* b3 = (const float*)d_in[6];
  const float* Wg = (const float*)d_in[7];
  const float* bg = (const float*)d_in[8];
  const float* gu = (const float*)d_in[9];
  float* out = (float*)d_out;

  char* ws = (char*)d_ws;
  int* counts = (int*)ws;
  int* idx    = (int*)(ws + 256);
  size_t off = 256 + (size_t)N_EXP * N_TOK * sizeof(int);
  off = (off + 255) & ~(size_t)255;
  const size_t wbytes = (size_t)N_EXP * H_DIM * H_DIM * sizeof(__bf16);
  __bf16* W1t = (__bf16*)(ws + off); off += wbytes;
  __bf16* W3t = (__bf16*)(ws + off); off += wbytes;
  __bf16* W2t = (__bf16*)(ws + off); off += wbytes;
  __bf16* xb     = (__bf16*)(ws + off); off += (size_t)N_TOK * H_DIM * 2;
  __bf16* innerb = (__bf16*)(ws + off);

  hipMemsetAsync(counts, 0, 256, stream);

  cvt_transpose_w<<<4096, 256, 0, stream>>>(W1, W1t);
  cvt_transpose_w<<<4096, 256, 0, stream>>>(W3, W3t);
  cvt_transpose_w<<<4096, 256, 0, stream>>>(W2, W2t);

  gate_route<<<N_TOK / 8, 256, 0, stream>>>(x, Wg, bg, gu, counts, idx, xb);

  dim3 grid(H_DIM / BN, N_TOK / BM, N_EXP);
  glu_up<<<grid, 256, 0, stream>>>(xb, W1t, W3t, b1, b3, counts, idx, innerb);
  glu_down<<<grid, 256, 0, stream>>>(innerb, W2t, b2, counts, idx, out);
}
